// Attention_42683384987651
// MI455X (gfx1250) — matrix operand order repaired and checked by reading, not compiled, not run on hardware
//
#include <hip/hip_runtime.h>

#define B_   2
#define L_   2048
#define D_   2048
#define H_   32
#define KV_  8
#define HD_  64
#define REP_ 4
#define KD_  (KV_ * HD_)   // 512

typedef __attribute__((ext_vector_type(16))) __bf16 v16bf;
typedef __attribute__((ext_vector_type(8)))  __bf16 v8bf;
typedef __attribute__((ext_vector_type(8)))  float  v8f;
typedef __attribute__((ext_vector_type(4)))  unsigned int u32x4;
typedef __attribute__((ext_vector_type(8)))  int  i32x8;
typedef __attribute__((ext_vector_type(4)))  int  i32x4;

#if defined(__has_builtin)
#if __has_builtin(__builtin_amdgcn_tensor_load_to_lds) && \
    __has_builtin(__builtin_amdgcn_s_wait_tensorcnt)
#define USE_TDM 1
#endif
#endif
#ifndef USE_TDM
#define USE_TDM 0
#endif

static __device__ __forceinline__ v16bf join16(v8bf lo, v8bf hi) {
  union { v16bf v; v8bf h[2]; } u;
  u.h[0] = lo; u.h[1] = hi;
  return u.v;
}
static __device__ __forceinline__ v8f zero8() {
  v8f z;
#pragma unroll
  for (int i = 0; i < 8; ++i) z[i] = 0.0f;
  return z;
}

#if USE_TDM
// Issue one TDM 2D tile copy: global (row-major, row stride = strideElems bf16,
// tile = 32 cols x 128 rows) -> LDS at byte offset lds_off.
// D# packing per CDNA5 ISA.
// This toolchain's builtin is the 6-arg form (clang-23 / therock-10.0):
// (g0:v4u, g1:v8i, g2:v4i, g3:v4i, extra:v8i, cpol:i32).
static __device__ __forceinline__ void tdm_tile_128x32(unsigned lds_off,
                                                       const __bf16* gptr,
                                                       int strideElems) {
  const unsigned long long ga = (unsigned long long)(uintptr_t)gptr;
  u32x4 g0;
  g0[0] = 1u;                                   // count=1, user mode
  g0[1] = lds_off;                              // lds_addr (bytes)
  g0[2] = (unsigned)ga;                         // global_addr[31:0]
  g0[3] = (unsigned)(ga >> 32) | 0x80000000u;   // global_addr[56:32] | type=2

  i32x8 g1;
  g1[0] = 0x00010000;                           // data_size=1 (2 bytes), no flags
  g1[1] = (int)((strideElems & 0xFFFF) << 16);  // tensor_dim0[15:0]
  g1[2] = (int)((((unsigned)strideElems >> 16) & 0xFFFFu) | (128u << 16)); // dim0 hi | tensor_dim1 lo
  g1[3] = (int)(32u << 16);                     // tensor_dim1 hi=0 | tile_dim0=32
  g1[4] = 128;                                  // tile_dim1=128, tile_dim2=0
  g1[5] = strideElems;                          // tensor_dim0_stride[31:0]
  g1[6] = 0;                                    // stride0 hi | stride1 lo
  g1[7] = 0;                                    // stride1 hi

  i32x4 gz4;
  gz4[0] = 0; gz4[1] = 0; gz4[2] = 0; gz4[3] = 0;  // 2D tensor: groups 2/3 unused
  i32x8 gz8;
#pragma unroll
  for (int j = 0; j < 8; ++j) gz8[j] = 0;
  __builtin_amdgcn_tensor_load_to_lds(g0, g1, gz4, gz4, gz8, 0);
}
#endif

// ---------------------------------------------------------------- fp32->bf16
__global__ __launch_bounds__(256)
void cvt_f32_bf16(const float* __restrict__ in, __bf16* __restrict__ out, int n4) {
  int i = blockIdx.x * 256 + threadIdx.x;
  if (i >= n4) return;
  float4 f = ((const float4*)in)[i];
  union { __bf16 h[4]; uint2 u; } t;
  t.h[0] = (__bf16)f.x; t.h[1] = (__bf16)f.y;
  t.h[2] = (__bf16)f.z; t.h[3] = (__bf16)f.w;
  ((uint2*)out)[i] = t.u;
}

// ------------------------------------------------- C(MxN) = A(MxK) @ W(NxK)^T
// Block tile 128x128, 8 waves (2x4), wave tile 64x32 -> 8 WMMA accumulators.
// A/W tiles staged to LDS by the Tensor Data Mover, double-buffered so the
// next tile's DMA overlaps the current tile's 8 WMMAs per K-step.
template <typename OutT>
__global__ __launch_bounds__(256)
void gemm_bf16(const __bf16* __restrict__ A, const __bf16* __restrict__ W,
               OutT* __restrict__ C, int M, int N, int K) {
  __shared__ __align__(16) __bf16 As[2][128 * 32];
  __shared__ __align__(16) __bf16 Bs[2][128 * 32];

  const int tid  = threadIdx.x;
  const int lane = tid & 31;
  const int l15  = lane & 15;
  const int lhi  = lane >> 4;       // half-wave select per ISA VGPR layouts
  const int wid  = tid >> 5;
  const int wm   = wid >> 2;        // 0..1
  const int wn   = wid & 3;         // 0..3
  const long bm  = (long)blockIdx.y * 128;
  const long bn  = (long)blockIdx.x * 128;

  v8f acc[4][2];
#pragma unroll
  for (int mi = 0; mi < 4; ++mi)
#pragma unroll
    for (int ni = 0; ni < 2; ++ni) acc[mi][ni] = zero8();

  const __bf16* Abase = A + bm * (long)K;
  const __bf16* Wbase = W + bn * (long)K;
  const int nit = K >> 5;           // K / 32

#if USE_TDM
  const unsigned ldsA0 = (unsigned)(uintptr_t)&As[0][0];
  const unsigned ldsA1 = (unsigned)(uintptr_t)&As[1][0];
  const unsigned ldsB0 = (unsigned)(uintptr_t)&Bs[0][0];
  const unsigned ldsB1 = (unsigned)(uintptr_t)&Bs[1][0];

  if (wid == 0) {                   // wave 0 drives the DMA engine
    tdm_tile_128x32(ldsA0, Abase, K);
    tdm_tile_128x32(ldsB0, Wbase, K);
  }
#else
  const int lrow = tid >> 1;
  const int lcol = (tid & 1) * 16;
#endif

  for (int i = 0; i < nit; ++i) {
    const int cur = i & 1;
    const int k0  = i << 5;

#if USE_TDM
    if (wid == 0) __builtin_amdgcn_s_wait_tensorcnt(0);  // cur tile landed
    __syncthreads();                                     // visible to all waves
    if (wid == 0 && (i + 1) < nit) {                     // overlap next DMA
      tdm_tile_128x32(cur ? ldsA0 : ldsA1, Abase + (long)(k0 + 32), K);
      tdm_tile_128x32(cur ? ldsB0 : ldsB1, Wbase + (long)(k0 + 32), K);
    }
#else
    __syncthreads();
    {
      const __bf16* ag = Abase + (long)lrow * K + k0 + lcol;
      const __bf16* wg = Wbase + (long)lrow * K + k0 + lcol;
      *(uint4*)&As[cur][lrow * 32 + lcol]     = *(const uint4*)ag;
      *(uint4*)&As[cur][lrow * 32 + lcol + 8] = *(const uint4*)(ag + 8);
      *(uint4*)&Bs[cur][lrow * 32 + lcol]     = *(const uint4*)wg;
      *(uint4*)&Bs[cur][lrow * 32 + lcol + 8] = *(const uint4*)(wg + 8);
      if (k0 + 32 < K) {
        __builtin_prefetch(ag + 32, 0, 0);
        __builtin_prefetch(wg + 32, 0, 0);
      }
    }
    __syncthreads();
#endif

    // A frag: lane holds row (l15), K = {kb..kb+7, kb+16..kb+23}, kb = 8*lhi
    v16bf af[4];
#pragma unroll
    for (int mi = 0; mi < 4; ++mi) {
      const __bf16* p = &As[cur][(wm * 64 + mi * 16 + l15) * 32 + lhi * 8];
      af[mi] = join16(*(const v8bf*)p, *(const v8bf*)(p + 16));
    }
    // B frag: lane holds col (l15), K = {kb..kb+7, kb+16..kb+23}, kb = 8*lhi
    v16bf bfr[2];
#pragma unroll
    for (int ni = 0; ni < 2; ++ni) {
      const __bf16* p = &Bs[cur][(wn * 32 + ni * 16 + l15) * 32 + lhi * 8];
      bfr[ni] = join16(*(const v8bf*)p, *(const v8bf*)(p + 16));
    }
#pragma unroll
    for (int mi = 0; mi < 4; ++mi)
#pragma unroll
      for (int ni = 0; ni < 2; ++ni)
        acc[mi][ni] = __builtin_amdgcn_wmma_f32_16x16x32_bf16(
            false, af[mi], false, bfr[ni], (short)0, acc[mi][ni], false, false);

    __syncthreads();                // all reads of `cur` done before reuse
  }

  // C layout: VGPR r, lanes0-15 -> row r, lanes16-31 -> row r+8; col = lane&15
#pragma unroll
  for (int mi = 0; mi < 4; ++mi)
#pragma unroll
    for (int ni = 0; ni < 2; ++ni) {
      const long col = bn + wn * 32 + ni * 16 + l15;
#pragma unroll
      for (int r = 0; r < 8; ++r) {
        const long row = bm + wm * 64 + mi * 16 + r + 8 * lhi;
        C[row * (long)N + col] = (OutT)acc[mi][ni][r];
      }
    }
}

// ---------------------------------------------------------------------- RoPE
// t layout (B, L, nheads, 64); interleaved pairs (2p, 2p+1)
__global__ __launch_bounds__(256)
void rope_bf16(__bf16* __restrict__ t, const float* __restrict__ cs,
               const float* __restrict__ sn, int nheads, int npairs) {
  int i = blockIdx.x * 256 + threadIdx.x;
  if (i >= npairs) return;
  const int p    = i & 31;
  const int pos  = (i / (32 * nheads)) % L_;
  const long base = (long)(i >> 5) * 64;
  const float xr = (float)t[base + 2 * p];
  const float xi = (float)t[base + 2 * p + 1];
  const float c = cs[pos * 32 + p];
  const float s = sn[pos * 32 + p];
  t[base + 2 * p]     = (__bf16)(xr * c - xi * s);
  t[base + 2 * p + 1] = (__bf16)(xr * s + xi * c);
}

// ------------------------------------------------------------ flash attention
// grid (L/64, B*H); block = 4 waves; wave = 16 query rows; key blocks of 32.
__global__ __launch_bounds__(128)
void flash_attn(const __bf16* __restrict__ Q, const __bf16* __restrict__ Kg,
                const __bf16* __restrict__ V, __bf16* __restrict__ O) {
  __shared__ __align__(16) __bf16 Ks[32 * 64];       // [key][hd]
  __shared__ __align__(16) __bf16 Vt[64 * 32];       // [hd][key] (transposed)
  __shared__ __align__(16) __bf16 Pb[4][16 * 32];    // per-wave P staging

  const int tid  = threadIdx.x;
  const int lane = tid & 31;
  const int l15  = lane & 15;
  const int lhi  = lane >> 4;
  const int w    = tid >> 5;

  const int bh = blockIdx.y;
  const int b  = bh / H_;
  const int h  = bh % H_;
  const int g  = h / REP_;            // kv head for this q head
  const int qt = blockIdx.x;
  const int qbase = qt * 64 + w * 16;

  const float SCALE = 0.125f;         // 1/sqrt(64)

  // Q fragments (A layout), K split into two 32-chunks of HD
  v16bf aq[2];
  {
    const long qrow = (long)b * L_ + qbase + l15;
#pragma unroll
    for (int c = 0; c < 2; ++c) {
      const __bf16* p = Q + qrow * D_ + h * HD_ + c * 32 + lhi * 8;
      aq[c] = join16(*(const v8bf*)p, *(const v8bf*)(p + 16));
    }
  }

  float m[8], lsum[8];
  v8f o[4];
#pragma unroll
  for (int r = 0; r < 8; ++r) { m[r] = -1e30f; lsum[r] = 0.0f; }
#pragma unroll
  for (int t = 0; t < 4; ++t) o[t] = zero8();

  const int nkb = qt * 2 + 2;           // same trip count for all 4 waves
  const int skey = tid >> 2;            // 0..31
  const int soff = (tid & 3) * 16;      // 0/16/32/48

  for (int kb = 0; kb < nkb; ++kb) {
    __syncthreads();
    {   // stage K (row-major) and V (transposed) to LDS
      const long krow = (long)b * L_ + kb * 32 + skey;
      const __bf16* kp = Kg + krow * KD_ + g * HD_ + soff;
      v8bf k0 = *(const v8bf*)kp;
      v8bf k1 = *(const v8bf*)(kp + 8);
      *(v8bf*)&Ks[skey * 64 + soff]     = k0;
      *(v8bf*)&Ks[skey * 64 + soff + 8] = k1;
      const __bf16* vp = V + krow * KD_ + g * HD_ + soff;
      v8bf v0 = *(const v8bf*)vp;
      v8bf v1 = *(const v8bf*)(vp + 8);
#pragma unroll
      for (int j = 0; j < 8; ++j) {
        Vt[(soff + j) * 32 + skey]     = v0[j];
        Vt[(soff + 8 + j) * 32 + skey] = v1[j];
      }
    }
    __syncthreads();

    // S(16x32) = Q @ K^T ; two 16x16 N tiles, K-dim = 64 in two WMMA steps
    v8f s0 = zero8(), s1 = zero8();
#pragma unroll
    for (int c = 0; c < 2; ++c) {
      const __bf16* p0 = &Ks[l15 * 64 + c * 32 + lhi * 8];
      const __bf16* p1 = &Ks[(16 + l15) * 64 + c * 32 + lhi * 8];
      v16bf bk0 = join16(*(const v8bf*)p0, *(const v8bf*)(p0 + 16));
      v16bf bk1 = join16(*(const v8bf*)p1, *(const v8bf*)(p1 + 16));
      s0 = __builtin_amdgcn_wmma_f32_16x16x32_bf16(false, aq[c], false, bk0, (short)0, s0, false, false);
      s1 = __builtin_amdgcn_wmma_f32_16x16x32_bf16(false, aq[c], false, bk1, (short)0, s1, false, false);
    }

    // online softmax: row r+8*lhi lives across the 16 lanes of a half-wave
#pragma unroll
    for (int r = 0; r < 8; ++r) {
      const int row = qbase + r + 8 * lhi;
      const int c0  = kb * 32 + l15;
      float v0 = (c0      <= row) ? s0[r] * SCALE : -1e30f;
      float v1 = (c0 + 16 <= row) ? s1[r] * SCALE : -1e30f;
      float mx = fmaxf(v0, v1);
#pragma unroll
      for (int off = 8; off >= 1; off >>= 1)
        mx = fmaxf(mx, __shfl_xor(mx, off, 16));
      const float mnew = fmaxf(m[r], mx);
      const float ef = __expf(m[r] - mnew);
      const float p0 = __expf(v0 - mnew);
      const float p1 = __expf(v1 - mnew);
      float sum = p0 + p1;
#pragma unroll
      for (int off = 8; off >= 1; off >>= 1)
        sum += __shfl_xor(sum, off, 16);
      lsum[r] = lsum[r] * ef + sum;
      m[r] = mnew;
      // C-layout -> LDS (row-major P tile) for A-layout reload
      Pb[w][(r + 8 * lhi) * 32 + l15]      = (__bf16)p0;
      Pb[w][(r + 8 * lhi) * 32 + l15 + 16] = (__bf16)p1;
#pragma unroll
      for (int t = 0; t < 4; ++t) o[t][r] *= ef;
    }

    // O(16x64) += P(16x32) @ V(32x64)
    const __bf16* pp = &Pb[w][l15 * 32 + lhi * 8];
    v16bf pa = join16(*(const v8bf*)pp, *(const v8bf*)(pp + 16));
#pragma unroll
    for (int t = 0; t < 4; ++t) {
      const __bf16* vp2 = &Vt[(t * 16 + l15) * 32 + lhi * 8];
      v16bf vb = join16(*(const v8bf*)vp2, *(const v8bf*)(vp2 + 16));
      o[t] = __builtin_amdgcn_wmma_f32_16x16x32_bf16(false, pa, false, vb, (short)0, o[t], false, false);
    }
  }

  // normalize and store (B, L, H, HD) as bf16 for the final projection
#pragma unroll
  for (int r = 0; r < 8; ++r) {
    const float inv = 1.0f / lsum[r];
    const long row = (long)b * L_ + qbase + r + 8 * lhi;
#pragma unroll
    for (int t = 0; t < 4; ++t)
      O[row * D_ + h * HD_ + t * 16 + l15] = (__bf16)(o[t][r] * inv);
  }
}

// ----------------------------------------------------------------- launcher
extern "C" void kernel_launch(void* const* d_in, const int* in_sizes, int n_in,
                              void* d_out, int out_size, void* d_ws, size_t ws_size,
                              hipStream_t stream) {
  (void)in_sizes; (void)n_in; (void)out_size; (void)ws_size;
  const float* x    = (const float*)d_in[0];
  const float* cosp = (const float*)d_in[1];
  const float* sinp = (const float*)d_in[2];
  const float* Wq   = (const float*)d_in[3];
  const float* Wk   = (const float*)d_in[4];
  const float* Wv   = (const float*)d_in[5];
  const float* Wo   = (const float*)d_in[6];
  float* out = (float*)d_out;

  const long nx  = (long)B_ * L_ * D_;   // 8388608
  const long nwq = (long)D_ * D_;        // 4194304
  const long nwk = (long)KD_ * D_;       // 1048576
  const long nk  = (long)B_ * L_ * KD_;  // 2097152

  char* w = (char*)d_ws;
  __bf16* xbf  = (__bf16*)w; w += nx  * 2;
  __bf16* wqbf = (__bf16*)w; w += nwq * 2;
  __bf16* wkbf = (__bf16*)w; w += nwk * 2;
  __bf16* wvbf = (__bf16*)w; w += nwk * 2;
  __bf16* wobf = (__bf16*)w; w += nwq * 2;
  __bf16* qbf  = (__bf16*)w; w += nx  * 2;
  __bf16* kbf  = (__bf16*)w; w += nk  * 2;
  __bf16* vbf  = (__bf16*)w; w += nk  * 2;
  __bf16* abf  = (__bf16*)w; w += nx  * 2;

  auto cvt = [&](const float* src, __bf16* dst, long n) {
    int n4 = (int)(n / 4);
    cvt_f32_bf16<<<dim3((n4 + 255) / 256), dim3(256), 0, stream>>>(src, dst, n4);
  };
  cvt(x, xbf, nx);
  cvt(Wq, wqbf, nwq);
  cvt(Wk, wkbf, nwk);
  cvt(Wv, wvbf, nwk);
  cvt(Wo, wobf, nwq);

  const int M = B_ * L_;
  gemm_bf16<__bf16><<<dim3(D_ / 128,  M / 128), 256, 0, stream>>>(xbf, wqbf, qbf, M, D_,  D_);
  gemm_bf16<__bf16><<<dim3(KD_ / 128, M / 128), 256, 0, stream>>>(xbf, wkbf, kbf, M, KD_, D_);
  gemm_bf16<__bf16><<<dim3(KD_ / 128, M / 128), 256, 0, stream>>>(xbf, wvbf, vbf, M, KD_, D_);

  rope_bf16<<<dim3((B_ * L_ * H_  * 32 + 255) / 256), 256, 0, stream>>>(qbf, cosp, sinp, H_,  B_ * L_ * H_  * 32);
  rope_bf16<<<dim3((B_ * L_ * KV_ * 32 + 255) / 256), 256, 0, stream>>>(kbf, cosp, sinp, KV_, B_ * L_ * KV_ * 32);

  flash_attn<<<dim3(L_ / 64, B_ * H_), 128, 0, stream>>>(qbf, kbf, vbf, abf);

  gemm_bf16<float><<<dim3(D_ / 128, M / 128), 256, 0, stream>>>(abf, wobf, out, M, D_, D_);
}
